// Seg_Strings_69741678953240
// MI455X (gfx1250) — matrix operand order repaired and checked by reading, not compiled, not run on hardware
//
#include <hip/hip_runtime.h>

typedef __attribute__((ext_vector_type(16))) _Float16 v16h;
typedef __attribute__((ext_vector_type(8)))  _Float16 v8h;
typedef __attribute__((ext_vector_type(8)))  float    v8f;

#define TPB 256
#define MAXK 70   // 70 slots * 5 feats = 350; padded row stride = 352 (K mult of 32)

static __device__ __forceinline__ int imin(int a, int b){ return a < b ? a : b; }

// ---------------- weight pack: f32 [K,N] row-major -> f16 WMMA B-fragment layout ----------------
// layout: dst[((ntile*ksteps + ks)*32 + lane)*16 + e]
//   lane<16 : col = ntile*16 + lane,    K = ks*32 + (e<8 ? e : 8 + e)
//   lane>=16: col = ntile*16 + lane-16, K = ks*32 + 8 + (e<8 ? e : 8 + e)
__global__ void pack_w_kernel(const float* __restrict__ W, _Float16* __restrict__ dst,
                              int K, int N, int ksteps) {
  int tid = blockIdx.x * blockDim.x + threadIdx.x;
  int total = (N / 16) * ksteps * 32;
  if (tid >= total) return;
  int lane = tid & 31;
  int ks   = (tid >> 5) % ksteps;
  int nt   = tid / (32 * ksteps);
  int col  = nt * 16 + (lane & 15);
  int kbase = ks * 32 + (lane >> 4) * 8;
  _Float16* d = dst + (size_t)tid * 16;
  for (int e = 0; e < 16; ++e) {
    int k = kbase + (e < 8 ? e : e + 8);
    d[e] = (k < K) ? (_Float16)W[(size_t)k * N + col] : (_Float16)0.0f;
  }
}

// ---------------- stack_bind scatter: rank via per-component counter, write 5 f16 feats ----------------
__global__ void scatter_kernel(const float* __restrict__ feat, const int* __restrict__ cc,
                               int Nitems, int* __restrict__ counters,
                               _Float16* __restrict__ xpack) {
  int i = blockIdx.x * blockDim.x + threadIdx.x;
  if (i >= Nitems) return;
  int c = cc[i];
  int r = atomicAdd(&counters[c], 1);
  if (r < MAXK) {
    size_t base = (size_t)c * 352 + (size_t)r * 5;
    const float* f = feat + (size_t)i * 5;
    for (int j = 0; j < 5; ++j) xpack[base + j] = (_Float16)f[j];
  }
}

// ---------------- WMMA GEMM: D[M,N] = act(A)[M,Kpad] * Bpack ----------------
// AF16: A is packed f16 (layer 1, already zero-padded). else: A is f32, fused
// BN+ReLU via per-column (a,b): act = max(0, z*a + b), converted to f16.
// Each wave: one 16-row stripe x NTW adjacent 16-col tiles; K-loop of 32.
template<bool AF16, int NTW>
__global__ __launch_bounds__(256) void gemm_wmma_kernel(
    const void* __restrict__ Asrc, const _Float16* __restrict__ Bp,
    float* __restrict__ D, int M, int Kpad, int N, int lda,
    const float* __restrict__ bna, const float* __restrict__ bnb) {
  int wg   = blockIdx.x * (blockDim.x >> 5) + (threadIdx.x >> 5);
  int lane = threadIdx.x & 31;
  int ntg    = N / (16 * NTW);
  int mtiles = (M + 15) >> 4;
  if (wg >= mtiles * ntg) return;
  int tileM  = wg / ntg;
  int tileN0 = (wg % ntg) * NTW;
  int sel = lane >> 4;        // 0: K chunk {0..7,16..23}; 1: {8..15,24..31}
  int rr  = lane & 15;        // row within tile (A), col within tile (B/D)
  int rowc = imin(tileM * 16 + rr, M - 1);
  int ksteps = Kpad >> 5;

  v8f acc[NTW];
  for (int nt = 0; nt < NTW; ++nt)
    for (int i = 0; i < 8; ++i) acc[nt][i] = 0.0f;

  for (int ks = 0; ks < ksteps; ++ks) {
    int k0 = ks * 32 + sel * 8;
    v16h av;
    if (AF16) {
      const _Float16* ar = (const _Float16*)Asrc + (size_t)rowc * lda;
      v8h lo = *(const v8h*)(ar + k0);
      v8h hi = *(const v8h*)(ar + k0 + 16);
      for (int j = 0; j < 8; ++j) { av[j] = lo[j]; av[j + 8] = hi[j]; }
    } else {
      const float* ar = (const float*)Asrc + (size_t)rowc * lda;
      for (int j = 0; j < 8; ++j) {
        int ka = k0 + j, kb = k0 + 16 + j;
        av[j]     = (_Float16)fmaxf(0.0f, fmaf(ar[ka], bna[ka], bnb[ka]));
        av[j + 8] = (_Float16)fmaxf(0.0f, fmaf(ar[kb], bna[kb], bnb[kb]));
      }
    }
    for (int nt = 0; nt < NTW; ++nt) {
      int tn = tileN0 + nt;
      const v16h bv = *(const v16h*)(Bp + ((size_t)(tn * ksteps + ks) * 32 + lane) * 16);
      acc[nt] = __builtin_amdgcn_wmma_f32_16x16x32_f16(
          false, av, false, bv, (short)0, acc[nt], false, false);
    }
  }
  // D layout: lane<16 -> col=lane, vgpr v -> row v; lane>=16 -> col=lane-16, row v+8
  for (int nt = 0; nt < NTW; ++nt) {
    int col = (tileN0 + nt) * 16 + rr;
    for (int v = 0; v < 8; ++v) {
      int orow = tileM * 16 + sel * 8 + v;
      if (orow < M) D[(size_t)orow * N + col] = acc[nt][v];
    }
  }
}

// ---------------- per-column sum / sum-of-squares over C rows ----------------
__global__ void col_stats_kernel(const float* __restrict__ Z, int C, int F, int RPB,
                                 float* __restrict__ sums, float* __restrict__ sumsq) {
  __shared__ float shs[TPB], shq[TPB];
  int tid = threadIdx.x;
  int col = tid % F;
  int rsub = tid / F;
  int stride = TPB / F;
  int r0 = blockIdx.x * RPB;
  int r1 = imin(r0 + RPB, C);
  float s = 0.f, q = 0.f;
  for (int r = r0 + rsub; r < r1; r += stride) {
    float v = Z[(size_t)r * F + col];
    s += v; q += v * v;
  }
  shs[tid] = s; shq[tid] = q;
  __syncthreads();
  for (int off = TPB >> 1; off >= F; off >>= 1) {
    if (tid < off) { shs[tid] += shs[tid + off]; shq[tid] += shq[tid + off]; }
    __syncthreads();
  }
  if (tid < F) { atomicAdd(&sums[tid], shs[tid]); atomicAdd(&sumsq[tid], shq[tid]); }
}

__global__ void bn_finalize_kernel(const float* __restrict__ sums, const float* __restrict__ sumsq,
                                   const float* __restrict__ g, const float* __restrict__ be,
                                   float* __restrict__ a, float* __restrict__ b, int F, float invC) {
  int t = blockIdx.x * blockDim.x + threadIdx.x;
  if (t >= F) return;
  float mean = sums[t] * invC;
  float var  = sumsq[t] * invC - mean * mean;
  float inv  = rsqrtf(var + 1e-5f);
  float aa   = g[t] * inv;
  a[t] = aa;
  b[t] = be[t] - mean * aa;
}

// ---------------- segment-sum pool: relu(bn(z4)) summed per image, LDS-staged ----------------
__global__ void pool_kernel(const float* __restrict__ z4, const int* __restrict__ imgid,
                            int C, int CPB, int B,
                            const float* __restrict__ a4, const float* __restrict__ b4,
                            float* __restrict__ pool) {
  __shared__ float acc[64 * 128];           // 32 KB: 64 images x 128 cols
  int t = threadIdx.x;                      // 0..127 -> one column
  int col = blockIdx.y * 128 + t;
  for (int i = t; i < 64 * 128; i += 128) acc[i] = 0.f;
  __syncthreads();
  int c0 = blockIdx.x * CPB;
  int c1 = imin(c0 + CPB, C);
  float av = a4[col], bv = b4[col];
  for (int c = c0; c < c1; ++c) {
    int img = imgid[c];
    float v = fmaxf(0.f, fmaf(z4[(size_t)c * 256 + col], av, bv));
    acc[img * 128 + t] += v;                // thread-private column: no races
  }
  __syncthreads();
  for (int img = 0; img < B; ++img)
    atomicAdd(&pool[(size_t)img * 256 + col], acc[img * 128 + t]);
}

// ---------------- pooled mean, BN over batch dim (single block, col per thread) ----------------
__global__ void final_bn_kernel(const float* __restrict__ pool, const int* __restrict__ imgid,
                                int C, int B, const float* __restrict__ g5,
                                const float* __restrict__ be5, float* __restrict__ x5n) {
  __shared__ int cnt[64];
  int t = threadIdx.x;
  if (t < 64) cnt[t] = 0;
  __syncthreads();
  for (int c = t; c < C; c += TPB) atomicAdd(&cnt[imgid[c]], 1);
  __syncthreads();
  float s = 0.f, q = 0.f;
  for (int img = 0; img < B; ++img) {
    float ct = (float)(cnt[img] < 1 ? 1 : cnt[img]);
    float x = pool[(size_t)img * 256 + t] / ct;
    s += x; q += x * x;
  }
  float mean = s / (float)B;
  float var  = q / (float)B - mean * mean;
  float inv  = rsqrtf(var + 1e-5f);
  float aa = g5[t] * inv;
  float bb = be5[t] - mean * aa;
  for (int img = 0; img < B; ++img) {
    float ct = (float)(cnt[img] < 1 ? 1 : cnt[img]);
    float x = pool[(size_t)img * 256 + t] / ct;
    x5n[(size_t)img * 256 + t] = x * aa + bb;
  }
}

// ---------------- FC 256 -> NC with bias + relu (tiny: 64 x 1000) ----------------
__global__ void fc_kernel(const float* __restrict__ x5n, const float* __restrict__ Wfc,
                          const float* __restrict__ bfc, float* __restrict__ out, int NC) {
  int j = blockIdx.x * blockDim.x + threadIdx.x;
  int img = blockIdx.y;
  if (j >= NC) return;
  float acc = bfc[j];
  const float* xr = x5n + (size_t)img * 256;
  for (int c = 0; c < 256; ++c) acc = fmaf(xr[c], Wfc[(size_t)c * NC + j], acc);
  out[(size_t)img * NC + j] = fmaxf(acc, 0.f);
}

extern "C" void kernel_launch(void* const* d_in, const int* in_sizes, int n_in,
                              void* d_out, int out_size, void* d_ws, size_t ws_size,
                              hipStream_t stream) {
  (void)n_in; (void)ws_size;
  const float* item_feat = (const float*)d_in[0];
  const int*   ccids     = (const int*)d_in[1];
  const int*   imgid     = (const int*)d_in[2];
  const float* W1  = (const float*)d_in[3];
  const float* g1  = (const float*)d_in[5];
  const float* be1 = (const float*)d_in[6];
  const float* W2  = (const float*)d_in[7];
  const float* g2  = (const float*)d_in[9];
  const float* be2 = (const float*)d_in[10];
  const float* W3  = (const float*)d_in[11];
  const float* g3  = (const float*)d_in[13];
  const float* be3 = (const float*)d_in[14];
  const float* W4  = (const float*)d_in[15];
  const float* g4  = (const float*)d_in[17];
  const float* be4 = (const float*)d_in[18];
  const float* g5  = (const float*)d_in[19];
  const float* be5 = (const float*)d_in[20];
  const float* Wfc = (const float*)d_in[21];
  const float* bfc = (const float*)d_in[22];
  // b1..b4 are algebraically absorbed by batch-stat BatchNorm -> unused.

  int Nitems = in_sizes[0] / 5;
  int C  = in_sizes[2];
  int NC = in_sizes[22];
  int B  = out_size / NC;
  float* out = (float*)d_out;

  // ---- workspace layout (z4 aliases dead xpack region) ----
  char* p = (char*)d_ws;
  size_t o = 0;
  auto take = [&](size_t bytes) -> char* {
    char* q = p + o; o = (o + bytes + 255) & ~(size_t)255; return q;
  };
  char* region0 = take((size_t)C * 1024);       // xpack f16 [C,352] (704B/row) then z4 f32 [C,256]
  _Float16* xpack = (_Float16*)region0;
  float*    z4    = (float*)region0;
  float* z1 = (float*)take((size_t)C * 32 * 4);
  float* z2 = (float*)take((size_t)C * 64 * 4);
  float* z3 = (float*)take((size_t)C * 128 * 4);
  int* counters = (int*)take((size_t)C * 4);
  _Float16* wp1 = (_Float16*)take(352 * 32 * 2);
  _Float16* wp2 = (_Float16*)take(32 * 64 * 2);
  _Float16* wp3 = (_Float16*)take(64 * 128 * 2);
  _Float16* wp4 = (_Float16*)take(128 * 256 * 2);
  float* stats = (float*)take(4 * 1024 * 4);    // per layer: sums|sumsq|a|b (256 each)
  float* pool  = (float*)take((size_t)B * 256 * 4);
  float* x5n   = (float*)take((size_t)B * 256 * 4);
  float* st0 = stats, *st1 = stats + 1024, *st2 = stats + 2048, *st3 = stats + 3072;

  hipMemsetAsync(xpack,    0, (size_t)C * 704, stream);
  hipMemsetAsync(counters, 0, (size_t)C * 4,   stream);
  hipMemsetAsync(stats,    0, 4 * 1024 * 4,    stream);
  hipMemsetAsync(pool,     0, (size_t)B * 256 * 4, stream);

  auto cdiv = [](long a, long b) { return (int)((a + b - 1) / b); };

  // pack weights into WMMA B-fragment layout (K zero-padded for W1: 350 -> 352)
  pack_w_kernel<<<cdiv(2L * 11 * 32, TPB), TPB, 0, stream>>>(W1, wp1, 350, 32, 11);
  pack_w_kernel<<<cdiv(4L * 1 * 32,  TPB), TPB, 0, stream>>>(W2, wp2, 32, 64, 1);
  pack_w_kernel<<<cdiv(8L * 2 * 32,  TPB), TPB, 0, stream>>>(W3, wp3, 64, 128, 2);
  pack_w_kernel<<<cdiv(16L * 4 * 32, TPB), TPB, 0, stream>>>(W4, wp4, 128, 256, 4);

  scatter_kernel<<<cdiv(Nitems, TPB), TPB, 0, stream>>>(item_feat, ccids, Nitems, counters, xpack);

  int mtiles = (C + 15) / 16;
  auto gblocks = [&](int ntg) { return cdiv((long)mtiles * ntg, 8); };

  // L1: [C,352]f16 @ [352,32] -> z1
  gemm_wmma_kernel<true, 2><<<gblocks(1), TPB, 0, stream>>>(
      xpack, wp1, z1, C, 352, 32, 352, nullptr, nullptr);
  col_stats_kernel<<<cdiv(C, 512), TPB, 0, stream>>>(z1, C, 32, 512, st0, st0 + 256);
  bn_finalize_kernel<<<1, TPB, 0, stream>>>(st0, st0 + 256, g1, be1, st0 + 512, st0 + 768, 32, 1.0f / (float)C);

  // L2: relu(bn(z1)) @ [32,64] -> z2
  gemm_wmma_kernel<false, 4><<<gblocks(1), TPB, 0, stream>>>(
      z1, wp2, z2, C, 32, 64, 32, st0 + 512, st0 + 768);
  col_stats_kernel<<<cdiv(C, 512), TPB, 0, stream>>>(z2, C, 64, 512, st1, st1 + 256);
  bn_finalize_kernel<<<1, TPB, 0, stream>>>(st1, st1 + 256, g2, be2, st1 + 512, st1 + 768, 64, 1.0f / (float)C);

  // L3: relu(bn(z2)) @ [64,128] -> z3
  gemm_wmma_kernel<false, 4><<<gblocks(2), TPB, 0, stream>>>(
      z2, wp3, z3, C, 64, 128, 64, st1 + 512, st1 + 768);
  col_stats_kernel<<<cdiv(C, 512), TPB, 0, stream>>>(z3, C, 128, 512, st2, st2 + 256);
  bn_finalize_kernel<<<1, TPB, 0, stream>>>(st2, st2 + 256, g3, be3, st2 + 512, st2 + 768, 128, 1.0f / (float)C);

  // L4: relu(bn(z3)) @ [128,256] -> z4 (aliases dead xpack)
  gemm_wmma_kernel<false, 4><<<gblocks(4), TPB, 0, stream>>>(
      z3, wp4, z4, C, 128, 256, 128, st2 + 512, st2 + 768);
  col_stats_kernel<<<cdiv(C, 512), TPB, 0, stream>>>(z4, C, 256, 512, st3, st3 + 256);
  bn_finalize_kernel<<<1, TPB, 0, stream>>>(st3, st3 + 256, g4, be4, st3 + 512, st3 + 768, 256, 1.0f / (float)C);

  // pool relu(bn(z4)) per image
  dim3 pg(cdiv(C, 2048), 2);
  pool_kernel<<<pg, 128, 0, stream>>>(z4, imgid, C, 2048, B, st3 + 512, st3 + 768, pool);

  final_bn_kernel<<<1, TPB, 0, stream>>>(pool, imgid, C, B, g5, be5, x5n);
  fc_kernel<<<dim3(cdiv(NC, TPB), B), TPB, 0, stream>>>(x5n, Wfc, bfc, out, NC);
}
